// PredictiveCodingNet_26482768347747
// MI455X (gfx1250) — matrix operand order repaired and checked by reading, not compiled, not run on hardware
//
#include <hip/hip_runtime.h>
#include <hip/hip_bf16.h>
#include <math.h>

// ---------------------------------------------------------------------------
// PredictiveCodingNet on MI455X (gfx1250): bf16 WMMA everywhere, x tile held
// in LDS across all 5 steps, h1/h2/h3 kept in WMMA C-fragments in registers.
// All weights pre-converted to bf16 in d_ws; small weights + ALL biases are
// staged in LDS so the step loop touches global memory only for fb1/W1.
// ---------------------------------------------------------------------------

typedef __attribute__((ext_vector_type(16))) __bf16 v16bf;
typedef __attribute__((ext_vector_type(8)))  float  v8f;

struct alignas(16) bf16x8 { __bf16 e[8]; };

#define XLD   792   // padded bf16 row stride for the x tile (bank-friendly)
#define SCRLD 36    // padded f32 row stride for per-wave transpose scratch

// ws (bf16 element offsets); each offset is a multiple of 8 -> 16B aligned
constexpr int OFF_FB1 = 0;            // fb1_w [784][128]
constexpr int OFF_W1  = 100352;       // W1_w  [128][800]  (cols 784..799 = 0)
constexpr int OFF_U1  = 202752;       // U1_w  [128][128]
constexpr int OFF_FB2 = 219136;       // fb2_w [128][64]
constexpr int OFF_W2  = 227328;       // W2_w  [64][128]
constexpr int OFF_U2  = 235520;       // U2_w  [64][64]
constexpr int OFF_FB3 = 239616;       // fb3_w [64][32]
constexpr int OFF_W3  = 241664;       // W3_w  [32][64]
constexpr int OFF_U3  = 243712;       // U3_w  [32][32]
constexpr int OFF_CLF = 244736;       // clf_w [16][32]    (rows 10..15 = 0)
constexpr int OFF_END = 245248;
constexpr int SMALL_BASE = OFF_U1;
constexpr int SMALL_CNT  = OFF_END - SMALL_BASE;   // 42496 bf16 = 84,992 B

// bias block in LDS (f32 element offsets)
constexpr int BB_FB1 = 0;      // 784
constexpr int BB_W1  = 784;    // 128
constexpr int BB_U1  = 912;    // 128
constexpr int BB_FB2 = 1040;   // 128
constexpr int BB_W2  = 1168;   // 64
constexpr int BB_U2  = 1232;   // 64
constexpr int BB_FB3 = 1296;   // 64
constexpr int BB_W3  = 1360;   // 32
constexpr int BB_U3  = 1392;   // 32
constexpr int BB_CLF = 1424;   // 16 (rows 10..15 = 0)
constexpr int BB_END = 1440;

__device__ __forceinline__ void lds_fence() {
  // wave-internal LDS write->read ordering (cross-lane through scratch);
  // DS ops are in-order per wave, the asm is a compiler + hw fence.
  asm volatile("s_wait_dscnt 0x0" ::: "memory");
}

__device__ __forceinline__ v8f vzero() {
  v8f c;
#pragma unroll
  for (int i = 0; i < 8; i++) c[i] = 0.0f;
  return c;
}

__device__ __forceinline__ v8f wmma_bf16(v16bf a, v16bf b, v8f c) {
  return __builtin_amdgcn_wmma_f32_16x16x32_bf16(false, a, false, b,
                                                 (short)0, c, false, false);
}

// B-fragment (32x16 = W^T tile) from row-major weight W[Nout][Kin]:
// lane (r=l&15, hi=l>>4) holds W[n0+r][k0 + hi*8 + 0..7] and W[n0+r][k0 + 16 + hi*8 + 0..7].
__device__ __forceinline__ v16bf load_bfrag(const __bf16* w, int ld, int n0,
                                            int k0, int r, int hi) {
  const __bf16* p = w + (size_t)(n0 + r) * ld + k0 + hi * 8;
  bf16x8 c0 = *(const bf16x8*)(p);
  bf16x8 c1 = *(const bf16x8*)(p + 16);
  v16bf out;
#pragma unroll
  for (int i = 0; i < 8; i++) { out[i] = c0.e[i]; out[i + 8] = c1.e[i]; }
  return out;
}

// D-fragment (16x16 f32, c[v] = D[v+hi*8][r]) -> scratch[M][n0 + r]
__device__ __forceinline__ void dfrag_to_scr(float* scr, v8f c, int n0,
                                             int r, int hi) {
  float* p = scr + hi * 8 * SCRLD + n0 + r;
#pragma unroll
  for (int v = 0; v < 8; v++) p[v * SCRLD] = c[v];
}

// A-fragment (16x32 bf16) from f32 scratch rows; ISA A layout:
// element e of lane (r,hi) is column ((e&8)*2 + hi*8 + (e&7)) of row r.
__device__ __forceinline__ v16bf afrag_scr(const float* scr, int r, int hi) {
  v16bf a;
  const float* p = scr + r * SCRLD;
#pragma unroll
  for (int e = 0; e < 16; e++) {
    int col = ((e & 8) << 1) + hi * 8 + (e & 7);
    a[e] = (__bf16)p[col];
  }
  return a;
}

// err_x A-fragment: bf16(x - pred), zero-padded beyond K=784
__device__ __forceinline__ v16bf afrag_err(const __bf16* xw, const float* scr,
                                           int k0, int r, int hi) {
  v16bf a;
#pragma unroll
  for (int e = 0; e < 16; e++) {
    int col = ((e & 8) << 1) + hi * 8 + (e & 7);
    int g = k0 + col;
    float v = 0.0f;
    if (g < 784) v = (float)xw[r * XLD + g] - scr[r * SCRLD + col];
    a[e] = (__bf16)v;
  }
  return a;
}

// two 16x16 C-fragments (cols 0..31) -> one 16x32 bf16 A-fragment
__device__ __forceinline__ v16bf transpose2(float* scr, v8f c0, v8f c1,
                                            int r, int hi) {
  dfrag_to_scr(scr, c0, 0, r, hi);
  dfrag_to_scr(scr, c1, 16, r, hi);
  lds_fence();
  v16bf a = afrag_scr(scr, r, hi);
  lds_fence();
  return a;
}

// ---------------------------------------------------------------------------
__global__ __launch_bounds__(256, 1) void pcnet_kernel(
    const float* __restrict__ x, const __bf16* __restrict__ wb,
    const float* __restrict__ W1_b, const float* __restrict__ U1_b,
    const float* __restrict__ W2_b, const float* __restrict__ U2_b,
    const float* __restrict__ W3_b, const float* __restrict__ U3_b,
    const float* __restrict__ fb3_b, const float* __restrict__ fb2_b,
    const float* __restrict__ fb1_b, const float* __restrict__ clf_b,
    const int* __restrict__ steps_p, float* __restrict__ out) {
  __shared__ __bf16 xs[128 * XLD];          // 202,752 B: x tile (all steps)
  __shared__ __bf16 wsm[SMALL_CNT];         //  84,992 B: small weights
  __shared__ float  scrbuf[8 * 16 * SCRLD]; //  18,432 B: per-wave transpose
  __shared__ float  bl[BB_END];             //   5,760 B: all biases

  const int tid = threadIdx.x;
  const int lane = tid & 31;
  const int wv = tid >> 5;
  const int r = lane & 15;
  const int hi = lane >> 4;

  // ---- stage x tile (f32 -> bf16, float4 loads) ----
  const float4* x4 = (const float4*)(x + (size_t)blockIdx.x * 128 * 784);
  for (int i = tid; i < 128 * 196; i += 256) {
    int row = i / 196;
    int c4 = i - row * 196;
    float4 v = x4[row * 196 + c4];
    __bf16* d = &xs[row * XLD + c4 * 4];
    d[0] = (__bf16)v.x; d[1] = (__bf16)v.y;
    d[2] = (__bf16)v.z; d[3] = (__bf16)v.w;
  }
  // ---- stage small weights (contiguous bf16 block) ----
  {
    const bf16x8* s = (const bf16x8*)(wb + SMALL_BASE);
    bf16x8* d = (bf16x8*)wsm;
    for (int i = tid; i < SMALL_CNT / 8; i += 256) d[i] = s[i];
  }
  // ---- stage all biases into LDS ----
  for (int i = tid; i < 784; i += 256) bl[BB_FB1 + i] = fb1_b[i];
  if (tid < 128) {
    bl[BB_W1 + tid]  = W1_b[tid];
    bl[BB_U1 + tid]  = U1_b[tid];
    bl[BB_FB2 + tid] = fb2_b[tid];
  }
  if (tid < 64) {
    bl[BB_W2 + tid]  = W2_b[tid];
    bl[BB_U2 + tid]  = U2_b[tid];
    bl[BB_FB3 + tid] = fb3_b[tid];
  }
  if (tid < 32) {
    bl[BB_W3 + tid] = W3_b[tid];
    bl[BB_U3 + tid] = U3_b[tid];
  }
  if (tid < 16) bl[BB_CLF + tid] = (tid < 10) ? clf_b[tid] : 0.0f;
  __syncthreads();

  const __bf16* fb1g = wb + OFF_FB1;
  const __bf16* w1g  = wb + OFF_W1;
  const __bf16* u1l  = wsm + (OFF_U1  - SMALL_BASE);
  const __bf16* fb2l = wsm + (OFF_FB2 - SMALL_BASE);
  const __bf16* w2l  = wsm + (OFF_W2  - SMALL_BASE);
  const __bf16* u2l  = wsm + (OFF_U2  - SMALL_BASE);
  const __bf16* fb3l = wsm + (OFF_FB3 - SMALL_BASE);
  const __bf16* w3l  = wsm + (OFF_W3  - SMALL_BASE);
  const __bf16* u3l  = wsm + (OFF_U3  - SMALL_BASE);
  const __bf16* clfl = wsm + (OFF_CLF - SMALL_BASE);

  float* scr = scrbuf + wv * 16 * SCRLD;
  const __bf16* xw = xs + wv * 16 * XLD;

  v8f h1f[8], h2f[4], h3f[2];
#pragma unroll
  for (int i = 0; i < 8; i++) h1f[i] = vzero();
#pragma unroll
  for (int i = 0; i < 4; i++) h2f[i] = vzero();
  h3f[0] = vzero(); h3f[1] = vzero();

  const int nsteps = *steps_p;
  for (int s = 0; s < nsteps; ++s) {
    // ================= layer 3: h3 += tanh(W3 err_h2) + tanh(U3 h3) ========
    v16bf a3 = transpose2(scr, h3f[0], h3f[1], r, hi);
    v8f accW3[2]; accW3[0] = vzero(); accW3[1] = vzero();
#pragma unroll
    for (int blk = 0; blk < 2; blk++) {   // err_h2 [16,64] in 32-col blocks
#pragma unroll
      for (int t = 0; t < 2; t++) {
        int n0 = blk * 32 + t * 16;
        v8f pred = wmma_bf16(a3, load_bfrag(fb3l, 32, n0, 0, r, hi), vzero());
        float b = bl[BB_FB3 + n0 + r];
        v8f e;
#pragma unroll
        for (int v = 0; v < 8; v++) e[v] = h2f[blk * 2 + t][v] - (pred[v] + b);
        dfrag_to_scr(scr, e, t * 16, r, hi);
      }
      lds_fence();
      v16bf ae = afrag_scr(scr, r, hi);
      lds_fence();
#pragma unroll
      for (int n = 0; n < 2; n++)
        accW3[n] = wmma_bf16(ae, load_bfrag(w3l, 64, n * 16, blk * 32, r, hi), accW3[n]);
    }
#pragma unroll
    for (int n = 0; n < 2; n++) {
      v8f accU = wmma_bf16(a3, load_bfrag(u3l, 32, n * 16, 0, r, hi), vzero());
      float bw = bl[BB_W3 + n * 16 + r], bu = bl[BB_U3 + n * 16 + r];
#pragma unroll
      for (int v = 0; v < 8; v++)
        h3f[n][v] += tanhf(accW3[n][v] + bw) + tanhf(accU[v] + bu);
    }

    // ================= layer 2 =============================================
    v16bf a2[2];
#pragma unroll
    for (int kb = 0; kb < 2; kb++)
      a2[kb] = transpose2(scr, h2f[2 * kb], h2f[2 * kb + 1], r, hi);
    v8f accW2[4];
#pragma unroll
    for (int n = 0; n < 4; n++) accW2[n] = vzero();
#pragma unroll
    for (int blk = 0; blk < 4; blk++) {   // err_h1 [16,128] in 32-col blocks
#pragma unroll
      for (int t = 0; t < 2; t++) {
        int n0 = blk * 32 + t * 16;
        v8f pred = vzero();
#pragma unroll
        for (int kb = 0; kb < 2; kb++)
          pred = wmma_bf16(a2[kb], load_bfrag(fb2l, 64, n0, kb * 32, r, hi), pred);
        float b = bl[BB_FB2 + n0 + r];
        v8f e;
#pragma unroll
        for (int v = 0; v < 8; v++) e[v] = h1f[blk * 2 + t][v] - (pred[v] + b);
        dfrag_to_scr(scr, e, t * 16, r, hi);
      }
      lds_fence();
      v16bf ae = afrag_scr(scr, r, hi);
      lds_fence();
#pragma unroll
      for (int n = 0; n < 4; n++)
        accW2[n] = wmma_bf16(ae, load_bfrag(w2l, 128, n * 16, blk * 32, r, hi), accW2[n]);
    }
#pragma unroll
    for (int n = 0; n < 4; n++) {
      v8f accU = vzero();
#pragma unroll
      for (int kb = 0; kb < 2; kb++)
        accU = wmma_bf16(a2[kb], load_bfrag(u2l, 64, n * 16, kb * 32, r, hi), accU);
      float bw = bl[BB_W2 + n * 16 + r], bu = bl[BB_U2 + n * 16 + r];
#pragma unroll
      for (int v = 0; v < 8; v++)
        h2f[n][v] += tanhf(accW2[n][v] + bw) + tanhf(accU[v] + bu);
    }

    // ================= layer 1 (the big one, K=784) ========================
    v16bf a1[4];
#pragma unroll
    for (int kb = 0; kb < 4; kb++)
      a1[kb] = transpose2(scr, h1f[2 * kb], h1f[2 * kb + 1], r, hi);
    v8f accW1[8];
#pragma unroll
    for (int n = 0; n < 8; n++) accW1[n] = vzero();
    for (int blk = 0; blk < 25; blk++) {  // err_x [16,784(pad 800)] blocks
#pragma unroll
      for (int t = 0; t < 2; t++) {
        int n0 = blk * 32 + t * 16;       // uniform: no EXEC divergence
        if (n0 < 784) {
          v8f pred = vzero();
#pragma unroll
          for (int kb = 0; kb < 4; kb++)
            pred = wmma_bf16(a1[kb], load_bfrag(fb1g, 128, n0, kb * 32, r, hi), pred);
          float b = bl[BB_FB1 + n0 + r];
#pragma unroll
          for (int v = 0; v < 8; v++) pred[v] += b;
          dfrag_to_scr(scr, pred, t * 16, r, hi);
        }
      }
      lds_fence();
      v16bf ae = afrag_err(xw, scr, blk * 32, r, hi);
      lds_fence();
#pragma unroll
      for (int n = 0; n < 8; n++)
        accW1[n] = wmma_bf16(ae, load_bfrag(w1g, 800, n * 16, blk * 32, r, hi), accW1[n]);
    }
#pragma unroll
    for (int n = 0; n < 8; n++) {
      v8f accU = vzero();
#pragma unroll
      for (int kb = 0; kb < 4; kb++)
        accU = wmma_bf16(a1[kb], load_bfrag(u1l, 128, n * 16, kb * 32, r, hi), accU);
      float bw = bl[BB_W1 + n * 16 + r], bu = bl[BB_U1 + n * 16 + r];
#pragma unroll
      for (int v = 0; v < 8; v++)
        h1f[n][v] += tanhf(accW1[n][v] + bw) + tanhf(accU[v] + bu);
    }
  }

  // ================= classifier: out = clf(h3) =============================
  {
    v16bf a3 = transpose2(scr, h3f[0], h3f[1], r, hi);
    v8f o = wmma_bf16(a3, load_bfrag(clfl, 32, 0, 0, r, hi), vzero());
    if (r < 10) {
      float b = bl[BB_CLF + r];
      size_t m0 = (size_t)blockIdx.x * 128 + wv * 16 + hi * 8;
#pragma unroll
      for (int v = 0; v < 8; v++) out[(m0 + v) * 10 + r] = o[v] + b;
    }
  }
}

// ---------------------------------------------------------------------------
// weight prep: f32 -> bf16 (with padding for W1 and clf)
__global__ void cvt_plain(const float* __restrict__ s, __bf16* __restrict__ d, int n) {
  int i = blockIdx.x * 256 + threadIdx.x;
  if (i < n) d[i] = (__bf16)s[i];
}
__global__ void cvt_w1pad(const float* __restrict__ s, __bf16* __restrict__ d) {
  int i = blockIdx.x * 256 + threadIdx.x;
  if (i < 128 * 800) {
    int row = i / 800, col = i - row * 800;
    d[i] = (col < 784) ? (__bf16)s[row * 784 + col] : (__bf16)0.0f;
  }
}
__global__ void cvt_clfpad(const float* __restrict__ s, __bf16* __restrict__ d) {
  int i = blockIdx.x * 256 + threadIdx.x;
  if (i < 16 * 32) {
    int row = i / 32, col = i - row * 32;
    d[i] = (row < 10) ? (__bf16)s[row * 32 + col] : (__bf16)0.0f;
  }
}

extern "C" void kernel_launch(void* const* d_in, const int* in_sizes, int n_in,
                              void* d_out, int out_size, void* d_ws, size_t ws_size,
                              hipStream_t stream) {
  const float* x     = (const float*)d_in[0];
  const float* W1_w  = (const float*)d_in[1];
  const float* W1_b  = (const float*)d_in[2];
  const float* U1_w  = (const float*)d_in[3];
  const float* U1_b  = (const float*)d_in[4];
  const float* W2_w  = (const float*)d_in[5];
  const float* W2_b  = (const float*)d_in[6];
  const float* U2_w  = (const float*)d_in[7];
  const float* U2_b  = (const float*)d_in[8];
  const float* W3_w  = (const float*)d_in[9];
  const float* W3_b  = (const float*)d_in[10];
  const float* U3_w  = (const float*)d_in[11];
  const float* U3_b  = (const float*)d_in[12];
  const float* fb3_w = (const float*)d_in[13];
  const float* fb3_b = (const float*)d_in[14];
  const float* fb2_w = (const float*)d_in[15];
  const float* fb2_b = (const float*)d_in[16];
  const float* fb1_w = (const float*)d_in[17];
  const float* fb1_b = (const float*)d_in[18];
  const float* clf_w = (const float*)d_in[19];
  const float* clf_b = (const float*)d_in[20];
  const int*   steps = (const int*)d_in[21];

  __bf16* wb = (__bf16*)d_ws;
  float* out = (float*)d_out;
  const int B = in_sizes[0] / 784;   // 65536

  auto blocks = [](int n) { return dim3((unsigned)((n + 255) / 256)); };
  cvt_plain<<<blocks(784 * 128), 256, 0, stream>>>(fb1_w, wb + OFF_FB1, 784 * 128);
  cvt_w1pad<<<blocks(128 * 800), 256, 0, stream>>>(W1_w, wb + OFF_W1);
  cvt_plain<<<blocks(128 * 128), 256, 0, stream>>>(U1_w, wb + OFF_U1, 128 * 128);
  cvt_plain<<<blocks(128 * 64), 256, 0, stream>>>(fb2_w, wb + OFF_FB2, 128 * 64);
  cvt_plain<<<blocks(64 * 128), 256, 0, stream>>>(W2_w, wb + OFF_W2, 64 * 128);
  cvt_plain<<<blocks(64 * 64), 256, 0, stream>>>(U2_w, wb + OFF_U2, 64 * 64);
  cvt_plain<<<blocks(64 * 32), 256, 0, stream>>>(fb3_w, wb + OFF_FB3, 64 * 32);
  cvt_plain<<<blocks(32 * 64), 256, 0, stream>>>(W3_w, wb + OFF_W3, 32 * 64);
  cvt_plain<<<blocks(32 * 32), 256, 0, stream>>>(U3_w, wb + OFF_U3, 32 * 32);
  cvt_clfpad<<<blocks(16 * 32), 256, 0, stream>>>(clf_w, wb + OFF_CLF);

  pcnet_kernel<<<dim3((unsigned)(B / 128)), dim3(256), 0, stream>>>(
      x, wb, W1_b, U1_b, W2_b, U2_b, W3_b, U3_b,
      fb3_b, fb2_b, fb1_b, clf_b, steps, out);
}
